// SelfAttention_71983651881305
// MI455X (gfx1250) — compile-verified, matrix operand order checked by reading, not run on hardware
//
#include <hip/hip_runtime.h>
#include <hip/hip_bf16.h>
#include <math.h>
#include <stdint.h>

#define HIDDEN 1024
#define BATCH  4
#define SEQ    2048

#define BK     128   // K-chunk staged per async copy
#define BROWS  64    // shared B rows per block
#define LDSW   136   // padded LDS row width (elements): 272B row stride

typedef __attribute__((ext_vector_type(16))) __bf16 v16bf;
typedef __attribute__((ext_vector_type(8)))  float  v8f;
typedef __attribute__((ext_vector_type(4)))  int    v4i;

// Address-space-qualified v4i for the async-to-LDS builtin:
// param0 = v4i addrspace(1)* (global src), param1 = v4i addrspace(3)* (LDS dst)
typedef __attribute__((address_space(1))) v4i gv4i;
typedef __attribute__((address_space(3))) v4i lv4i;

#define WMMA_BF16(a, b, c) \
  __builtin_amdgcn_wmma_f32_16x16x32_bf16(false, (a), false, (b), (short)0, (c), false, false)

// ---------------------------------------------------------------------------
// Async global -> LDS 16-byte copy (per lane), tracked by ASYNCcnt.
// ---------------------------------------------------------------------------
static __device__ __forceinline__ void async_b128(const void* g, void* l) {
#if __has_builtin(__builtin_amdgcn_global_load_async_to_lds_b128)
  __builtin_amdgcn_global_load_async_to_lds_b128((gv4i*)g, (lv4i*)l, 0, 0);
#else
  const unsigned lds32 = (unsigned)(uintptr_t)l;  // flat-LDS low 32 bits = LDS offset
  asm volatile("global_load_async_to_lds_b128 %0, %1, off"
               :: "v"(lds32), "v"((unsigned long long)(uintptr_t)g)
               : "memory");
#endif
}

template <int N>
static __device__ __forceinline__ void wait_async() {
#if __has_builtin(__builtin_amdgcn_s_wait_asynccnt)
  __builtin_amdgcn_s_wait_asynccnt(N);
#else
  asm volatile("s_wait_asynccnt %0" :: "i"(N) : "memory");
#endif
}

// ---------------------------------------------------------------------------
// Fragment loaders matching CDNA5 16-bit A/B VGPR striping:
// lanes 0-15 (M/N = lane):  K = kb+0..7 in v[0..7], kb+16..23 in v[8..15]
// lanes 16-31 (M/N = l-16): K = kb+8..15 and kb+24..31.
// ---------------------------------------------------------------------------
static __device__ __forceinline__ v16bf frag_from_f32(const float* __restrict__ row,
                                                      int kb, int lane) {
  const int ksub = (lane >> 4) << 3;
  const float* p0 = row + kb + ksub;
  const float* p1 = p0 + 16;
  v16bf f;
#pragma unroll
  for (int i = 0; i < 8; ++i) {
    f[i]     = (__bf16)p0[i];
    f[i + 8] = (__bf16)p1[i];
  }
  return f;
}

static __device__ __forceinline__ v16bf frag_from_bf16(const __bf16* __restrict__ row,
                                                       int kb, int lane) {
  const int ksub = (lane >> 4) << 3;
  const __bf16* p0 = row + kb + ksub;
  const __bf16* p1 = p0 + 16;
  v16bf f;
#pragma unroll
  for (int i = 0; i < 8; ++i) {
    f[i]     = p0[i];
    f[i + 8] = p1[i];
  }
  return f;
}

static __device__ __forceinline__ v16bf frag_from_lds(const __bf16* row, int kk, int lane) {
  const int ksub = (lane >> 4) << 3;
  const __bf16* p0 = row + kk + ksub;
  const __bf16* p1 = p0 + 16;
  v16bf f;
#pragma unroll
  for (int i = 0; i < 8; ++i) {
    f[i]     = p0[i];
    f[i + 8] = p1[i];
  }
  return f;
}

// Copy one BROWSxBK bf16 chunk (rows rowbase..rowbase+63, cols kb..kb+BK-1 of a
// [*, stride] bf16 matrix) into an LDS tile. 256 threads, 4 async b128 each.
static __device__ __forceinline__ void copy_chunk_bf16(__bf16 (*dst)[LDSW],
                                                       const __bf16* __restrict__ src,
                                                       int rowbase, int kb, size_t stride,
                                                       int crow, int ccol) {
#pragma unroll
  for (int i = 0; i < 4; ++i) {
    const int r = crow + i * 16;
    const __bf16* g = src + (size_t)(rowbase + r) * stride + kb + ccol;
    async_b128((const void*)g, (void*)&dst[r][ccol]);
  }
}

// =========================================================================
// Stage 1: Q/K/V projections.  y[m,o] = sum_h x[m,h]*W[o,h] + b[o]
// blockIdx.z: 0->Q, 1->K, 2->V (V stored transposed: Vt[b][h][s]).
// grid (64, 16, 3), block 256 (8 waves); wave strip 16 rows x 64 cols.
// =========================================================================
__global__ void __launch_bounds__(256) qkv_proj_kernel(
    const float* __restrict__ x,
    const float* __restrict__ Wq, const float* __restrict__ bq,
    const float* __restrict__ Wk, const float* __restrict__ bk,
    const float* __restrict__ Wv, const float* __restrict__ bv,
    __bf16* __restrict__ Qb, __bf16* __restrict__ Kb, __bf16* __restrict__ Vt) {
  const int lane  = threadIdx.x & 31;
  const int wave  = threadIdx.x >> 5;
  const int mbase = (blockIdx.x * 8 + wave) * 16;
  const int nbase = blockIdx.y * 64;
  const int which = blockIdx.z;

  const float* W    = (which == 0) ? Wq : (which == 1) ? Wk : Wv;
  const float* bias = (which == 0) ? bq : (which == 1) ? bk : bv;

  const float* arow  = x + (size_t)(mbase + (lane & 15)) * HIDDEN;
  const float* brow0 = W + (size_t)(nbase +  0 + (lane & 15)) * HIDDEN;
  const float* brow1 = W + (size_t)(nbase + 16 + (lane & 15)) * HIDDEN;
  const float* brow2 = W + (size_t)(nbase + 32 + (lane & 15)) * HIDDEN;
  const float* brow3 = W + (size_t)(nbase + 48 + (lane & 15)) * HIDDEN;

  v8f acc0 = {}, acc1 = {}, acc2 = {}, acc3 = {};
  for (int kb = 0; kb < HIDDEN; kb += 32) {
    __builtin_prefetch((const void*)(arow + kb + 128), 0, 1);
    v16bf a  = frag_from_f32(arow, kb, lane);
    v16bf b0 = frag_from_f32(brow0, kb, lane);
    v16bf b1 = frag_from_f32(brow1, kb, lane);
    v16bf b2 = frag_from_f32(brow2, kb, lane);
    v16bf b3 = frag_from_f32(brow3, kb, lane);
    acc0 = WMMA_BF16(a, b0, acc0);
    acc1 = WMMA_BF16(a, b1, acc1);
    acc2 = WMMA_BF16(a, b2, acc2);
    acc3 = WMMA_BF16(a, b3, acc3);
  }

  const int rowoff = (lane < 16) ? 0 : 8;
  const int ncol   = lane & 15;
  v8f accs[4] = {acc0, acc1, acc2, acc3};
#pragma unroll
  for (int t = 0; t < 4; ++t) {
#pragma unroll
    for (int r = 0; r < 8; ++r) {
      const int row = mbase + rowoff + r;
      const int col = nbase + t * 16 + ncol;
      const float v = accs[t][r] + bias[col];
      if (which == 0) {
        Qb[(size_t)row * HIDDEN + col] = (__bf16)v;
      } else if (which == 1) {
        Kb[(size_t)row * HIDDEN + col] = (__bf16)v;
      } else {
        const int b = row >> 11;
        const int s = row & (SEQ - 1);
        Vt[((size_t)b * HIDDEN + col) * SEQ + s] = (__bf16)v;
      }
    }
  }
}

// =========================================================================
// Stage 2: scores[b,q,k] = (Q[b,q,:].K[b,k,:]) / 32, masked to -inf.
// grid (16, 32, 4), block 256. 64 shared K-rows staged via async->LDS,
// double-buffered in BK=128 chunks; each wave computes a 16x64 strip.
// =========================================================================
__global__ void __launch_bounds__(256) score_kernel(
    const __bf16* __restrict__ Qb, const __bf16* __restrict__ Kb,
    const int* __restrict__ mask, float* __restrict__ Sc) {
  __shared__ __bf16 bt[2][BROWS][LDSW];

  const int tid   = threadIdx.x;
  const int lane  = tid & 31;
  const int wave  = tid >> 5;
  const int b     = blockIdx.z;
  const int qbase = (blockIdx.x * 8 + wave) * 16;
  const int kbase = blockIdx.y * BROWS;

  const __bf16* Q    = Qb + (size_t)b * SEQ * HIDDEN;
  const __bf16* K    = Kb + (size_t)b * SEQ * HIDDEN;
  const __bf16* qrow = Q + (size_t)(qbase + (lane & 15)) * HIDDEN;

  const int crow = tid >> 4;        // 0..15 (+16*i)
  const int ccol = (tid & 15) * 8;  // element column within chunk

  copy_chunk_bf16(bt[0], K, kbase, 0, HIDDEN, crow, ccol);

  v8f acc0 = {}, acc1 = {}, acc2 = {}, acc3 = {};
  const int NCH = HIDDEN / BK;  // 8
  for (int c = 0; c < NCH; ++c) {
    const int kb = c * BK;
    if (c + 1 < NCH) {
      copy_chunk_bf16(bt[(c + 1) & 1], K, kbase, kb + BK, HIDDEN, crow, ccol);
      wait_async<4>();
    } else {
      wait_async<0>();
    }
    __syncthreads();
    const __bf16 (*tile)[LDSW] = bt[c & 1];
#pragma unroll
    for (int kk = 0; kk < BK; kk += 32) {
      __builtin_prefetch((const void*)(qrow + kb + kk + 128), 0, 1);
      v16bf a  = frag_from_bf16(qrow, kb + kk, lane);
      v16bf b0 = frag_from_lds(&tile[ 0 + (lane & 15)][0], kk, lane);
      v16bf b1 = frag_from_lds(&tile[16 + (lane & 15)][0], kk, lane);
      v16bf b2 = frag_from_lds(&tile[32 + (lane & 15)][0], kk, lane);
      v16bf b3 = frag_from_lds(&tile[48 + (lane & 15)][0], kk, lane);
      acc0 = WMMA_BF16(a, b0, acc0);
      acc1 = WMMA_BF16(a, b1, acc1);
      acc2 = WMMA_BF16(a, b2, acc2);
      acc3 = WMMA_BF16(a, b3, acc3);
    }
    __syncthreads();
  }

  const float scale = 0.03125f;  // 1/sqrt(1024)
  const int* mrow   = mask + (size_t)b * SEQ * SEQ;
  float*     srow   = Sc   + (size_t)b * SEQ * SEQ;
  const int rowoff  = (lane < 16) ? 0 : 8;
  const int ncol    = lane & 15;
  v8f accs[4] = {acc0, acc1, acc2, acc3};
#pragma unroll
  for (int t = 0; t < 4; ++t) {
#pragma unroll
    for (int r = 0; r < 8; ++r) {
      const int q = qbase + rowoff + r;
      const int k = kbase + t * 16 + ncol;
      float v = accs[t][r] * scale;
      if (mrow[(size_t)q * SEQ + k] == 0) v = -__builtin_inff();
      srow[(size_t)q * SEQ + k] = v;
    }
  }
}

// =========================================================================
// Stage 3: row softmax over 2048 f32 scores -> bf16 probabilities.
// =========================================================================
__global__ void __launch_bounds__(256) softmax_kernel(const float* __restrict__ Sc,
                                                      __bf16* __restrict__ P) {
  __shared__ float red[256];
  const int row = blockIdx.x;
  const int tid = threadIdx.x;
  const float* s = Sc + (size_t)row * SEQ;

  float v[8];
  float m = -__builtin_inff();
#pragma unroll
  for (int i = 0; i < 8; ++i) { v[i] = s[tid + i * 256]; m = fmaxf(m, v[i]); }
  red[tid] = m; __syncthreads();
  for (int off = 128; off > 0; off >>= 1) {
    if (tid < off) red[tid] = fmaxf(red[tid], red[tid + off]);
    __syncthreads();
  }
  const float rowmax = red[0];
  __syncthreads();

  float sum = 0.0f;
#pragma unroll
  for (int i = 0; i < 8; ++i) { v[i] = __expf(v[i] - rowmax); sum += v[i]; }
  red[tid] = sum; __syncthreads();
  for (int off = 128; off > 0; off >>= 1) {
    if (tid < off) red[tid] += red[tid + off];
    __syncthreads();
  }
  const float inv = 1.0f / red[0];

  __bf16* p = P + (size_t)row * SEQ;
#pragma unroll
  for (int i = 0; i < 8; ++i) p[tid + i * 256] = (__bf16)(v[i] * inv);
}

// =========================================================================
// Stage 4: out[b,q,h] = sum_k P[b,q,k] * Vt[b,h,k].
// grid (16, 16, 4), block 256. 64 shared Vt-rows async-staged in LDS,
// K = SEQ = 2048 in BK=128 double-buffered chunks.
// =========================================================================
__global__ void __launch_bounds__(256) pv_kernel(
    const __bf16* __restrict__ P, const __bf16* __restrict__ Vt,
    float* __restrict__ out) {
  __shared__ __bf16 bt[2][BROWS][LDSW];

  const int tid   = threadIdx.x;
  const int lane  = tid & 31;
  const int wave  = tid >> 5;
  const int b     = blockIdx.z;
  const int qbase = (blockIdx.x * 8 + wave) * 16;
  const int hbase = blockIdx.y * BROWS;

  const __bf16* Pb   = P  + (size_t)b * SEQ * SEQ;
  const __bf16* Vb   = Vt + (size_t)b * HIDDEN * SEQ;
  const __bf16* prow = Pb + (size_t)(qbase + (lane & 15)) * SEQ;

  const int crow = tid >> 4;
  const int ccol = (tid & 15) * 8;

  copy_chunk_bf16(bt[0], Vb, hbase, 0, SEQ, crow, ccol);

  v8f acc0 = {}, acc1 = {}, acc2 = {}, acc3 = {};
  const int NCH = SEQ / BK;  // 16
  for (int c = 0; c < NCH; ++c) {
    const int kb = c * BK;
    if (c + 1 < NCH) {
      copy_chunk_bf16(bt[(c + 1) & 1], Vb, hbase, kb + BK, SEQ, crow, ccol);
      wait_async<4>();
    } else {
      wait_async<0>();
    }
    __syncthreads();
    const __bf16 (*tile)[LDSW] = bt[c & 1];
#pragma unroll
    for (int kk = 0; kk < BK; kk += 32) {
      __builtin_prefetch((const void*)(prow + kb + kk + 128), 0, 1);
      v16bf a  = frag_from_bf16(prow, kb + kk, lane);
      v16bf b0 = frag_from_lds(&tile[ 0 + (lane & 15)][0], kk, lane);
      v16bf b1 = frag_from_lds(&tile[16 + (lane & 15)][0], kk, lane);
      v16bf b2 = frag_from_lds(&tile[32 + (lane & 15)][0], kk, lane);
      v16bf b3 = frag_from_lds(&tile[48 + (lane & 15)][0], kk, lane);
      acc0 = WMMA_BF16(a, b0, acc0);
      acc1 = WMMA_BF16(a, b1, acc1);
      acc2 = WMMA_BF16(a, b2, acc2);
      acc3 = WMMA_BF16(a, b3, acc3);
    }
    __syncthreads();
  }

  float* orow = out + (size_t)b * SEQ * HIDDEN;
  const int rowoff = (lane < 16) ? 0 : 8;
  const int ncol   = lane & 15;
  v8f accs[4] = {acc0, acc1, acc2, acc3};
#pragma unroll
  for (int t = 0; t < 4; ++t) {
#pragma unroll
    for (int r = 0; r < 8; ++r) {
      const int q = qbase + rowoff + r;
      const int h = hbase + t * 16 + ncol;
      orow[(size_t)q * HIDDEN + h] = accs[t][r];
    }
  }
}

// =========================================================================
// Host launcher. Workspace layout (144 MB total):
//   [0,16M)     Q  bf16  [B*S, H]
//   [16M,32M)   K  bf16  [B*S, H]
//   [32M,48M)   Vt bf16  [B, H, S]
//   [48M,112M)  Sc f32   [B, S, S]
//   [112M,144M) P  bf16  [B, S, S]
// =========================================================================
extern "C" void kernel_launch(void* const* d_in, const int* in_sizes, int n_in,
                              void* d_out, int out_size, void* d_ws, size_t ws_size,
                              hipStream_t stream) {
  (void)in_sizes; (void)n_in; (void)out_size; (void)ws_size;
  const float* x    = (const float*)d_in[0];
  const int*   mask = (const int*)d_in[1];
  const float* Wq   = (const float*)d_in[2];
  const float* bq   = (const float*)d_in[3];
  const float* Wk   = (const float*)d_in[4];
  const float* bk   = (const float*)d_in[5];
  const float* Wv   = (const float*)d_in[6];
  const float* bv   = (const float*)d_in[7];
  float* out = (float*)d_out;

  char* ws = (char*)d_ws;
  __bf16* Qb = (__bf16*)(ws);
  __bf16* Kb = (__bf16*)(ws + ((size_t)16 << 20));
  __bf16* Vt = (__bf16*)(ws + ((size_t)32 << 20));
  float*  Sc = (float*) (ws + ((size_t)48 << 20));
  __bf16* P  = (__bf16*)(ws + ((size_t)112 << 20));

  dim3 blk(256);
  qkv_proj_kernel<<<dim3(64, 16, 3), blk, 0, stream>>>(x, Wq, bq, Wk, bk, Wv, bv, Qb, Kb, Vt);
  score_kernel<<<dim3(16, 32, 4), blk, 0, stream>>>(Qb, Kb, mask, Sc);
  softmax_kernel<<<dim3(BATCH * SEQ), blk, 0, stream>>>(Sc, P);
  pv_kernel<<<dim3(16, 16, 4), blk, 0, stream>>>(P, Vt, out);
}
